// AdaptiveFusion_42296837931312
// MI455X (gfx1250) — matrix operand order repaired and checked by reading, not compiled, not run on hardware
//
#include <hip/hip_runtime.h>
#include <math.h>

typedef __attribute__((ext_vector_type(16))) __bf16 v16bf;
typedef __attribute__((ext_vector_type(8)))  float  v8f;
typedef __attribute__((ext_vector_type(4)))  float  f32x4;   // native vector for nontemporal store
typedef unsigned short u16;
typedef unsigned int   u32;

// ---------------------------------------------------------------- helpers
__device__ __forceinline__ u16 f32_to_bf16(float f) {
    union { float f; u32 u; } c; c.f = f;
    u32 x = c.u;
    u32 r = x + 0x7FFFu + ((x >> 16) & 1u);   // round-to-nearest-even
    return (u16)(r >> 16);
}

union Frag { v16bf v; uint4 q[2]; };

// ---------------------------------------------------------------- fp32 -> bf16 convert
__global__ __launch_bounds__(256) void cvt_f32_to_bf16(const float* __restrict__ in,
                                                       u16* __restrict__ out, long n) {
    const long i = ((long)blockIdx.x * 256 + threadIdx.x) * 4;
    if (i >= n) return;
    const float4 v = *(const float4*)(in + i);
    union { u16 h[4]; uint2 u; } r;
    r.h[0] = f32_to_bf16(v.x); r.h[1] = f32_to_bf16(v.y);
    r.h[2] = f32_to_bf16(v.z); r.h[3] = f32_to_bf16(v.w);
    *(uint2*)(out + i) = r.u;
}

// ---------------------------------------------------------------- generic batched WMMA GEMM
// C[m,n] = scale * sum_h A_bf16[m,h] * W_bf16[n,h] + bias[n]  (optionally += existing C)
// Block = 128 threads = 4 waves; each wave computes a 16(M) x 32(N) tile.
// grid = (N/32, ceil(M/64), batch)
template <bool OUT_BF16, bool ACCUM>
__global__ __launch_bounds__(128) void gemm_bf16_wmma(
    const u16* __restrict__ A, long lda, long aBat,
    const u16* __restrict__ W, long ldw, long wBat,
    void* __restrict__ Cp, long ldc, long cBat,
    const float* __restrict__ bias, float scale, int M, int Kd)
{
    const int tid  = threadIdx.x;
    const int wave = tid >> 5, lane = tid & 31;
    const int lo = lane & 15, hi = lane >> 4;
    const int m0 = (blockIdx.y * 4 + wave) * 16;
    if (m0 >= M) return;
    const int n0 = blockIdx.x * 32;

    A += (size_t)blockIdx.z * aBat;
    W += (size_t)blockIdx.z * wBat;

    const u16* a0 = A + (size_t)(m0 + lo) * lda + hi * 16;       // K = k0 + hi*16 .. +7
    const u16* a1 = a0 + 8;                                      // K = k0+hi*16+8 .. +15
    const u16* w0 = W + (size_t)(n0 + lo) * ldw + hi * 16;       // K = k0 + hi*16 .. +15
    const u16* w1 = W + (size_t)(n0 + 16 + lo) * ldw + hi * 16;

    v8f acc0 = {}; v8f acc1 = {};
    for (int k0 = 0; k0 < Kd; k0 += 32) {
        Frag af, bf0, bf1;
        af.q[0]  = *(const uint4*)(a0 + k0);
        af.q[1]  = *(const uint4*)(a1 + k0);
        bf0.q[0] = *(const uint4*)(w0 + k0);
        bf0.q[1] = *(const uint4*)(w0 + k0 + 8);
        bf1.q[0] = *(const uint4*)(w1 + k0);
        bf1.q[1] = *(const uint4*)(w1 + k0 + 8);
        acc0 = __builtin_amdgcn_wmma_f32_16x16x32_bf16(false, af.v, false, bf0.v,
                                                       (short)0, acc0, false, false);
        acc1 = __builtin_amdgcn_wmma_f32_16x16x32_bf16(false, af.v, false, bf1.v,
                                                       (short)0, acc1, false, false);
    }

    const float b0v = bias ? bias[n0 + lo] : 0.0f;
    const float b1v = bias ? bias[n0 + 16 + lo] : 0.0f;

    if constexpr (OUT_BF16) {
        u16* C = (u16*)Cp + (size_t)blockIdx.z * cBat;
#pragma unroll
        for (int i = 0; i < 8; ++i) {
            const size_t r = (size_t)(m0 + hi * 8 + i) * ldc;
            C[r + n0 + lo]      = f32_to_bf16(acc0[i] * scale + b0v);
            C[r + n0 + 16 + lo] = f32_to_bf16(acc1[i] * scale + b1v);
        }
    } else {
        float* C = (float*)Cp + (size_t)blockIdx.z * cBat;
#pragma unroll
        for (int i = 0; i < 8; ++i) {
            const size_t r = (size_t)(m0 + hi * 8 + i) * ldc;
            float v0 = acc0[i] * scale + b0v;
            float v1 = acc1[i] * scale + b1v;
            if constexpr (ACCUM) {
                v0 += C[r + n0 + lo];
                v1 += C[r + n0 + 16 + lo];
            }
            C[r + n0 + lo]      = v0;
            C[r + n0 + 16 + lo] = v1;
        }
    }
}

// ---------------------------------------------------------------- softmax over 128 cols, emit bf16
// one wave per row; 4 values per lane; grid = rows/8, block = 256
__global__ __launch_bounds__(256) void softmax128_to_bf16(const float* __restrict__ S,
                                                          u16* __restrict__ Sb) {
    const int wave = threadIdx.x >> 5, lane = threadIdx.x & 31;
    const size_t row = (size_t)blockIdx.x * 8 + wave;
    const float4 v = *(const float4*)(S + row * 128 + lane * 4);
    float m = fmaxf(fmaxf(v.x, v.y), fmaxf(v.z, v.w));
#pragma unroll
    for (int off = 16; off; off >>= 1) m = fmaxf(m, __shfl_xor(m, off, 32));
    const float e0 = __expf(v.x - m), e1 = __expf(v.y - m);
    const float e2 = __expf(v.z - m), e3 = __expf(v.w - m);
    float sum = e0 + e1 + e2 + e3;
#pragma unroll
    for (int off = 16; off; off >>= 1) sum += __shfl_xor(sum, off, 32);
    const float inv = 1.0f / sum;
    union { u16 h[4]; uint2 u; } r;
    r.h[0] = f32_to_bf16(e0 * inv); r.h[1] = f32_to_bf16(e1 * inv);
    r.h[2] = f32_to_bf16(e2 * inv); r.h[3] = f32_to_bf16(e3 * inv);
    *(uint2*)(Sb + row * 128 + lane * 4) = r.u;
}

// ---------------------------------------------------------------- gate 1: h1 = sig(lg)*x + (1-sig)*gctx
__global__ __launch_bounds__(256) void gate1_epilogue(const float* __restrict__ lg,
                                                      const float* __restrict__ x,
                                                      const float* __restrict__ g,
                                                      float* __restrict__ h1,
                                                      u16* __restrict__ h1b, long n) {
    const long i = ((long)blockIdx.x * 256 + threadIdx.x) * 4;
    if (i >= n) return;
    const float4 l = *(const float4*)(lg + i);
    const float4 xv = *(const float4*)(x + i);
    const float4 gv = *(const float4*)(g + i);
    float4 h;
    float t;
    t = 1.0f / (1.0f + __expf(-l.x)); h.x = t * xv.x + (1.0f - t) * gv.x;
    t = 1.0f / (1.0f + __expf(-l.y)); h.y = t * xv.y + (1.0f - t) * gv.y;
    t = 1.0f / (1.0f + __expf(-l.z)); h.z = t * xv.z + (1.0f - t) * gv.z;
    t = 1.0f / (1.0f + __expf(-l.w)); h.w = t * xv.w + (1.0f - t) * gv.w;
    *(float4*)(h1 + i) = h;
    union { u16 hh[4]; uint2 u; } r;
    r.hh[0] = f32_to_bf16(h.x); r.hh[1] = f32_to_bf16(h.y);
    r.hh[2] = f32_to_bf16(h.z); r.hh[3] = f32_to_bf16(h.w);
    *(uint2*)(h1b + i) = r.u;
}

// ---------------------------------------------------------------- final streaming epilogue (201 MB out)
// out[b,l,k,o] = sig(A2[b,l,o] + B2[b,k,o]) * h1[b,l,o] + (1-sig) * s[b,k,o]
// (bg2 bias already folded into A2 by its GEMM)
__global__ __launch_bounds__(256) void final_epilogue(const float* __restrict__ A2,
                                                      const float* __restrict__ B2,
                                                      const float* __restrict__ h1,
                                                      const float* __restrict__ s,
                                                      float* __restrict__ out) {
    const size_t idx = (size_t)blockIdx.x * 256 + threadIdx.x;
    const int o = (int)(idx % 192) * 4;        // H/4 = 192
    size_t t = idx / 192;
    const int k = (int)(t & 31);  t >>= 5;     // K = 32
    const int l = (int)(t & 255);              // L = 256
    const int b = (int)(t >> 8);
    const size_t rowBL = (size_t)(b * 256 + l) * 768 + o;
    const size_t rowBK = (size_t)(b * 32 + k) * 768 + o;
    const float4 a2 = *(const float4*)(A2 + rowBL);
    const float4 b2 = *(const float4*)(B2 + rowBK);
    const float4 hv = *(const float4*)(h1 + rowBL);
    const float4 sv = *(const float4*)(s + rowBK);
    f32x4 r;
    float gt;
    gt = 1.0f / (1.0f + __expf(-(a2.x + b2.x))); r.x = gt * hv.x + (1.0f - gt) * sv.x;
    gt = 1.0f / (1.0f + __expf(-(a2.y + b2.y))); r.y = gt * hv.y + (1.0f - gt) * sv.y;
    gt = 1.0f / (1.0f + __expf(-(a2.z + b2.z))); r.z = gt * hv.z + (1.0f - gt) * sv.z;
    gt = 1.0f / (1.0f + __expf(-(a2.w + b2.w))); r.w = gt * hv.w + (1.0f - gt) * sv.w;
    __builtin_nontemporal_store(r, (f32x4*)(out + idx * 4));
}

// ---------------------------------------------------------------- host launcher
extern "C" void kernel_launch(void* const* d_in, const int* in_sizes, int n_in,
                              void* d_out, int out_size, void* d_ws, size_t ws_size,
                              hipStream_t stream) {
    (void)in_sizes; (void)n_in; (void)out_size; (void)ws_size;
    const float* x   = (const float*)d_in[0];
    const float* s   = (const float*)d_in[1];
    const float* g   = (const float*)d_in[2];
    const float* Wq  = (const float*)d_in[3];
    const float* bq  = (const float*)d_in[4];
    const float* Wkv = (const float*)d_in[5];
    const float* bkv = (const float*)d_in[6];
    const float* Wg1 = (const float*)d_in[7];
    const float* bg1 = (const float*)d_in[8];
    const float* Wg2 = (const float*)d_in[9];
    const float* bg2 = (const float*)d_in[10];

    constexpr long H = 768, B = 8, L = 256, K = 32, Lg = 128;
    constexpr long ML = B * L;   // 2048 rows
    constexpr long Mg = B * Lg;  // 1024 rows
    constexpr long Ms = B * K;   //  256 rows

    char* p = (char*)d_ws;
    auto alloc_u16 = [&](long n) { u16* r = (u16*)p; p += ((n * 2 + 255) & ~255L); return r; };
    auto alloc_f32 = [&](long n) { float* r = (float*)p; p += ((n * 4 + 255) & ~255L); return r; };

    u16* xb    = alloc_u16(ML * H);
    u16* gb    = alloc_u16(Mg * H);
    u16* sb    = alloc_u16(Ms * H);
    u16* Wqb   = alloc_u16(H * H);
    u16* Wkvb  = alloc_u16(2 * H * H);
    u16* Wg1b  = alloc_u16(H * 2 * H);
    u16* Wg2b  = alloc_u16(H * 2 * H);
    u16* qb    = alloc_u16(ML * H);
    u16* kvb   = alloc_u16(Mg * 2 * H);
    u16* scb   = alloc_u16(B * L * Lg);
    u16* gctxb = alloc_u16(ML * H);
    u16* h1b   = alloc_u16(ML * H);
    float* scores  = alloc_f32(B * L * Lg);
    float* gctx    = alloc_f32(ML * H);
    float* logits1 = alloc_f32(ML * H);
    float* h1      = alloc_f32(ML * H);
    float* A2      = alloc_f32(ML * H);
    float* B2      = alloc_f32(Ms * H);

    auto cvt = [&](const float* in, u16* o, long n) {
        cvt_f32_to_bf16<<<dim3((unsigned)((n / 4 + 255) / 256)), dim3(256), 0, stream>>>(in, o, n);
    };
    cvt(x, xb, ML * H);
    cvt(g, gb, Mg * H);
    cvt(s, sb, Ms * H);
    cvt(Wq, Wqb, H * H);
    cvt(Wkv, Wkvb, 2 * H * H);
    cvt(Wg1, Wg1b, H * 2 * H);
    cvt(Wg2, Wg2b, H * 2 * H);

    // q = x @ Wq^T + bq -> bf16
    gemm_bf16_wmma<true, false><<<dim3(H / 32, ML / 64, 1), dim3(128), 0, stream>>>(
        xb, H, 0, Wqb, H, 0, qb, H, 0, bq, 1.0f, (int)ML, (int)H);
    // kv = g @ Wkv^T + bkv -> bf16 (k = cols [0,H), v = cols [H,2H))
    gemm_bf16_wmma<true, false><<<dim3(2 * H / 32, Mg / 64, 1), dim3(128), 0, stream>>>(
        gb, H, 0, Wkvb, H, 0, kvb, 2 * H, 0, bkv, 1.0f, (int)Mg, (int)H);
    // scores = (q @ k^T) / sqrt(H), batched over B
    gemm_bf16_wmma<false, false><<<dim3(Lg / 32, L / 64, B), dim3(128), 0, stream>>>(
        qb, H, L * H, kvb, 2 * H, Lg * 2 * H, scores, Lg, L * Lg,
        nullptr, 0.03608439182435161f, (int)L, (int)H);
    // softmax over 128, emit bf16
    softmax128_to_bf16<<<dim3((unsigned)(ML / 8)), dim3(256), 0, stream>>>(scores, scb);
    // gctx = softmax(scores) @ v, batched over B
    gemm_bf16_wmma<false, false><<<dim3(H / 32, L / 64, B), dim3(128), 0, stream>>>(
        scb, Lg, L * Lg, kvb + H, 2 * H, Lg * 2 * H, gctx, H, L * H,
        nullptr, 1.0f, (int)L, (int)Lg);
    cvt(gctx, gctxb, ML * H);
    // logits1 = x @ Wx1^T + bg1 ; then += gctx @ Wy1^T
    gemm_bf16_wmma<false, false><<<dim3(H / 32, ML / 64, 1), dim3(128), 0, stream>>>(
        xb, H, 0, Wg1b, 2 * H, 0, logits1, H, 0, bg1, 1.0f, (int)ML, (int)H);
    gemm_bf16_wmma<false, true><<<dim3(H / 32, ML / 64, 1), dim3(128), 0, stream>>>(
        gctxb, H, 0, Wg1b + H, 2 * H, 0, logits1, H, 0, nullptr, 1.0f, (int)ML, (int)H);
    // gate 1 -> h1 (fp32 + bf16)
    gate1_epilogue<<<dim3((unsigned)(ML * H / 4 / 256)), dim3(256), 0, stream>>>(
        logits1, x, gctx, h1, h1b, ML * H);
    // A2 = h1 @ Wx2^T + bg2 ; B2 = s @ Wy2^T
    gemm_bf16_wmma<false, false><<<dim3(H / 32, ML / 64, 1), dim3(128), 0, stream>>>(
        h1b, H, 0, Wg2b, 2 * H, 0, A2, H, 0, bg2, 1.0f, (int)ML, (int)H);
    gemm_bf16_wmma<false, false><<<dim3(H / 32, Ms / 64, 1), dim3(128), 0, stream>>>(
        sb, H, 0, Wg2b + H, 2 * H, 0, B2, H, 0, nullptr, 1.0f, (int)Ms, (int)H);
    // final 201 MB streaming epilogue
    final_epilogue<<<dim3((unsigned)(B * L * K * H / 4 / 256)), dim3(256), 0, stream>>>(
        A2, B2, h1, s, (float*)d_out);
}
